// FourierAttention_28338194219709
// MI455X (gfx1250) — matrix operand order repaired and checked by reading, not compiled, not run on hardware
//
#include <hip/hip_runtime.h>
#include <math.h>

// ---------------------------------------------------------------------------
// FourierAttention on MI455X (gfx1250, wave32, WMMA bf16 -> f32 accum)
//
// Pipeline (FFTs folded into weights as linear maps on the d=128 axis):
//   K0a: W_ftT  [4608][128]  = fold(w_qkv, DFT basis)     (bf16, B-transposed)
//   K0b: W_invT [128][1536]  = fold(IDFT basis, w_out)    (bf16, B-transposed)
//   Kx : xbf    [32768][128] = bf16(x)   (one-time convert; stays L2-resident)
//   K1 : ft     [32768][4608] = x @ W_ft                  (WMMA bf16)
//   K1a: qkT    [2][b][h][96][2048] = |q_ft|,|k_ft| transposed (freq-major)
//   K2 : att_raw[bh][96][96] = qT @ kT^T (K=2048)         (WMMA bf16, 6-tile rows)
//   K3 : att    = softmax(att_raw) over 65, zero-padded to 96 (bf16)
//   K4 : xqkv   [32768][1536] = v_ft(re,im) @ att^T       (WMMA bf16)
//   K5 : out    [32768][128]  = xqkv @ W_inv + b_out      (WMMA bf16, f32 out)
// ---------------------------------------------------------------------------

typedef __attribute__((ext_vector_type(16))) __bf16 v16bf;
typedef __attribute__((ext_vector_type(8)))  __bf16 v8bf;
typedef __attribute__((ext_vector_type(8)))  float  v8f;
typedef unsigned short bf16_t;   // storage type (host-safe)

#define BB      16
#define NN      2048
#define DIMD    128
#define HEADS   8
#define NF      65
#define PF      96                   // padded frequency block
#define HV      (2*PF)               // 192 (re+im) per head
#define FT_COLS (3*HEADS*HV)         // 4608
#define XQ_COLS (HEADS*HV)           // 1536
#define MM      (BB*NN)              // 32768
#define SCALE_F (1.0f/65.0f)
#define TWO_PI  6.28318530717958647692f

// ------------------------------ bf16 helpers -------------------------------
// Native f32 -> bf16 conversion (lowers to hardware cvt on gfx1250; avoids
// the v_add3/v_and software-RNE sequences seen with manual rounding).
static __device__ __forceinline__ bf16_t f2bf(float f) {
  __bf16 h = (__bf16)f;
  return __builtin_bit_cast(bf16_t, h);
}
static __device__ __forceinline__ float bf2f(bf16_t b) {
  unsigned u = ((unsigned)b) << 16;
  return __builtin_bit_cast(float, u);
}

// ------------------------- WMMA fragment loaders ---------------------------
// A-matrix 16x32 bf16 as loaded here (contiguous k, not AMD's table order; B matches):
//   lanes 0-15 : M=lane,    K = k0..k0+15
//   lanes 16-31: M=lane-16, K = k0+16..k0+31
static __device__ __forceinline__ v16bf load_a_bf16(const bf16_t* base, int ld,
                                                    int m0, int k0) {
  int lane = threadIdx.x & 31;
  int m  = m0 + (lane & 15);
  int ko = k0 + ((lane >> 4) ? 16 : 0);
  const bf16_t* p = base + (size_t)m * ld + ko;
  v8bf lo = *(const v8bf*)(const void*)(p);
  v8bf hi = *(const v8bf*)(const void*)(p + 8);
  return __builtin_shufflevector(lo, hi, 0,1,2,3,4,5,6,7,8,9,10,11,12,13,14,15);
}

// B-matrix 32x16 bf16, operand stored TRANSPOSED (row = output column n,
// contiguous K):  lanes 0-15: col=lane, K=k0..k0+15 ; lanes 16-31: K=k0+16..k0+31
static __device__ __forceinline__ v16bf load_bT(const bf16_t* baseT, int ld,
                                                int n0, int k0) {
  int lane = threadIdx.x & 31;
  int n  = n0 + (lane & 15);
  int ko = k0 + ((lane >> 4) ? 16 : 0);
  const bf16_t* p = baseT + (size_t)n * ld + ko;
  v8bf lo = *(const v8bf*)(const void*)(p);
  v8bf hi = *(const v8bf*)(const void*)(p + 8);
  return __builtin_shufflevector(lo, hi, 0,1,2,3,4,5,6,7,8,9,10,11,12,13,14,15);
}

static __device__ __forceinline__ v8f wmma_bf16(v16bf a, v16bf b, v8f c) {
  // (neg_a, A, neg_b, B, c_mod, C, reuse_a, reuse_b)
  return __builtin_amdgcn_wmma_f32_16x16x32_bf16(false, a, false, b,
                                                 (short)0, c, false, false);
}

// ---------------------- K0a: fold w_qkv with DFT basis ---------------------
// W_ftT[col][d],  col = t*1536 + h*192 + c*96 + f  (c: 0=re, 1=im)
__global__ void k_fold_wft(const float* __restrict__ w_qkv,
                           bf16_t* __restrict__ wftT) {
  int id = blockIdx.x * blockDim.x + threadIdx.x;
  if (id >= FT_COLS * DIMD) return;
  int d   = id & (DIMD - 1);
  int col = id >> 7;
  int t = col / (HEADS * HV);
  int r = col % (HEADS * HV);
  int h = r / HV;  r %= HV;
  int c = r / PF;
  int f = r % PF;
  float acc = 0.f;
  if (f < NF) {
    float sc = (t < 2) ? SCALE_F : 1.0f;   // q,k scaled before rfft
    const float* w = w_qkv + (size_t)d * (3 * HEADS * DIMD) + t * HEADS * DIMD + h * DIMD;
    float w0 = TWO_PI * (float)f / 128.0f;
    for (int e = 0; e < DIMD; ++e) {
      float ang = w0 * (float)e;
      float bas = (c == 0) ? __cosf(ang) : -__sinf(ang);   // rfft: re, im
      acc += w[e] * bas;
    }
    acc *= sc;
  }
  wftT[(size_t)col * DIMD + d] = f2bf(acc);
}

// --------------------- K0b: fold IDFT basis with w_out ---------------------
// W_invT[d][k],  k = h*192 + c*96 + f
__global__ void k_fold_winv(const float* __restrict__ w_out,
                            bf16_t* __restrict__ winvT) {
  int id = blockIdx.x * blockDim.x + threadIdx.x;
  if (id >= DIMD * XQ_COLS) return;
  int k = id % XQ_COLS;
  int d = id / XQ_COLS;
  int h = k / HV; int r = k % HV;
  int c = r / PF; int f = r % PF;
  float acc = 0.f;
  if (f < NF) {
    const float* w = w_out + (size_t)(h * DIMD) * DIMD + d;   // w_out[h*128+e][d]
    float w0 = TWO_PI * (float)f / 128.0f;
    for (int e = 0; e < DIMD; ++e) {
      float coef;
      if (c == 0) {                                   // real part
        if (f == 0)        coef = 1.0f / 128.0f;
        else if (f == 64)  coef = ((e & 1) ? -1.0f : 1.0f) / 128.0f;
        else               coef = (2.0f / 128.0f) * __cosf(w0 * (float)e);
      } else {                                        // imag part
        coef = (f == 0 || f == 64) ? 0.0f
                                   : (-2.0f / 128.0f) * __sinf(w0 * (float)e);
      }
      acc += coef * w[(size_t)e * DIMD];
    }
  }
  winvT[(size_t)d * XQ_COLS + k] = f2bf(acc);
}

// -------------------- Kx: one-time x f32 -> bf16 staging -------------------
__global__ void k_x2bf(const float* __restrict__ x, bf16_t* __restrict__ xbf) {
  int id = blockIdx.x * blockDim.x + threadIdx.x;
  if (id < MM * DIMD) xbf[id] = f2bf(x[id]);
}

// ------------------- K1: ft = x @ W_ft  (M=32768,K=128,N=4608) -------------
// block = 8 waves; each wave: 16 M-rows x 64 N-cols (4 accumulators)
__global__ void __launch_bounds__(256)
k_gemm_ft(const bf16_t* __restrict__ xbf, const bf16_t* __restrict__ wftT,
          bf16_t* __restrict__ ft) {
  int wave = threadIdx.x >> 5;
  int m0 = blockIdx.x * 128 + wave * 16;
  int n0 = blockIdx.y * 64;
  v8f acc[4] = {};
  for (int k0 = 0; k0 < DIMD; k0 += 32) {
    v16bf a = load_a_bf16(xbf, DIMD, m0, k0);
#pragma unroll
    for (int j = 0; j < 4; ++j) {
      v16bf b = load_bT(wftT, DIMD, n0 + j * 16, k0);
      acc[j] = wmma_bf16(a, b, acc[j]);
    }
  }
  int lane = threadIdx.x & 31;
  int colb = n0 + (lane & 15);
#pragma unroll
  for (int j = 0; j < 4; ++j)
#pragma unroll
    for (int i = 0; i < 8; ++i) {
      int row = m0 + i + ((lane >> 4) ? 8 : 0);
      ft[(size_t)row * FT_COLS + colb + j * 16] = f2bf(acc[j][i]);
    }
}

// ----------- K1a: |q_ft|,|k_ft| -> transposed freq-major bf16 --------------
// qkT layout: [t][b][h][f(96)][n(2048)]
__global__ void k_abs_t(const bf16_t* __restrict__ ft,
                        bf16_t* __restrict__ qkT) {
  size_t id = (size_t)blockIdx.x * blockDim.x + threadIdx.x;
  if (id >= (size_t)2 * BB * HEADS * PF * NN) return;
  int n = (int)(id % NN); size_t r = id / NN;
  int f = (int)(r % PF);  r /= PF;
  int h = (int)(r % HEADS); r /= HEADS;
  int b = (int)(r % BB);
  int t = (int)(r / BB);
  float v = 0.f;
  if (f < NF) {
    size_t row  = (size_t)b * NN + n;
    size_t base = row * FT_COLS + (size_t)t * (HEADS * HV) + (size_t)h * HV;
    float re = bf2f(ft[base + f]);
    float im = bf2f(ft[base + PF + f]);
    v = sqrtf(re * re + im * im);
  }
  qkT[id] = f2bf(v);
}

// --------------- K2: att_raw[bh] = qT @ kT^T  (65x65, K=2048) --------------
// one wave per 16-row band: 6 accumulators, A-fragment reused 6x per K step
__global__ void __launch_bounds__(32)
k_gemm_att(const bf16_t* __restrict__ qkT, float* __restrict__ att_raw) {
  int bh = blockIdx.x / 6;
  int tx = blockIdx.x % 6;
  const bf16_t* qb = qkT + (size_t)bh * PF * NN;
  const bf16_t* kb = qkT + ((size_t)BB * HEADS + bh) * PF * NN;
  v8f acc[6] = {};
  for (int k0 = 0; k0 < NN; k0 += 32) {
    if (k0 + 32 < NN)
      __builtin_prefetch(qb + (size_t)(tx * 16) * NN + k0 + 32, 0, 0);
    v16bf a = load_a_bf16(qb, NN, tx * 16, k0);
#pragma unroll
    for (int j = 0; j < 6; ++j) {
      v16bf b = load_bT(kb, NN, j * 16, k0);
      acc[j] = wmma_bf16(a, b, acc[j]);
    }
  }
  int lane = threadIdx.x & 31;
  float* out = att_raw + (size_t)bh * PF * PF;
#pragma unroll
  for (int j = 0; j < 6; ++j) {
    int y = j * 16 + (lane & 15);
#pragma unroll
    for (int i = 0; i < 8; ++i) {
      int xr = tx * 16 + i + ((lane >> 4) ? 8 : 0);
      out[(size_t)xr * PF + y] = acc[j][i];
    }
  }
}

// ----------------------- K3: row softmax over 65, pad->0 -------------------
__global__ void __launch_bounds__(128)
k_softmax(const float* __restrict__ att_raw, bf16_t* __restrict__ att) {
  __shared__ float red[128];
  int x  = blockIdx.x % PF;
  int bh = blockIdx.x / PF;
  const float* in = att_raw + ((size_t)bh * PF + x) * PF;
  bf16_t* out = att + ((size_t)bh * PF + x) * PF;
  int tid = threadIdx.x;
  if (x >= NF) { if (tid < PF) out[tid] = 0; return; }
  float v = (tid < NF) ? in[tid] : -3.0e38f;
  red[tid] = v; __syncthreads();
  for (int s = 64; s > 0; s >>= 1) {
    if (tid < s) red[tid] = fmaxf(red[tid], red[tid + s]);
    __syncthreads();
  }
  float mx = red[0]; __syncthreads();
  float e = (tid < NF) ? __expf(v - mx) : 0.f;
  red[tid] = e; __syncthreads();
  for (int s = 64; s > 0; s >>= 1) {
    if (tid < s) red[tid] += red[tid + s];
    __syncthreads();
  }
  float inv = 1.f / red[0];
  if (tid < PF) out[tid] = f2bf((tid < NF) ? e * inv : 0.f);
}

// ------ K4: xqkv[.., h,c,x] = v_ft(c) @ att^T  (M=2048,K=96,N=96 / bh,c) ---
__global__ void __launch_bounds__(256)
k_gemm_xqkv(const bf16_t* __restrict__ ft, const bf16_t* __restrict__ att,
            bf16_t* __restrict__ xq) {
  int blk = blockIdx.x;
  int mb = blk & 15; int rest = blk >> 4;
  int c = rest & 1;  rest >>= 1;
  int h = rest & 7;  int b = rest >> 3;
  int wave = threadIdx.x >> 5;
  int nrow0 = mb * 128 + wave * 16;
  const bf16_t* abase = ft + (size_t)b * NN * FT_COLS
                           + 2 * (HEADS * HV) + h * HV + c * PF;
  const bf16_t* bbase = att + (size_t)(b * HEADS + h) * PF * PF;
  v8f acc[6] = {};
  for (int k0 = 0; k0 < PF; k0 += 32) {
    v16bf a = load_a_bf16(abase, FT_COLS, nrow0, k0);
#pragma unroll
    for (int j = 0; j < 6; ++j) {
      v16bf bfrag = load_bT(bbase, PF, j * 16, k0);
      acc[j] = wmma_bf16(a, bfrag, acc[j]);
    }
  }
  int lane = threadIdx.x & 31;
#pragma unroll
  for (int j = 0; j < 6; ++j) {
    int xcol = j * 16 + (lane & 15);
#pragma unroll
    for (int i = 0; i < 8; ++i) {
      int nrow = nrow0 + i + ((lane >> 4) ? 8 : 0);
      size_t row = (size_t)b * NN + nrow;
      xq[row * XQ_COLS + h * HV + c * PF + xcol] = f2bf(acc[j][i]);
    }
  }
}

// --------- K5: out = xqkv @ W_inv + b_out  (M=32768,K=1536,N=128) ----------
__global__ void __launch_bounds__(256)
k_gemm_out(const bf16_t* __restrict__ xq, const bf16_t* __restrict__ winvT,
           const float* __restrict__ b_out, float* __restrict__ out) {
  int wave = threadIdx.x >> 5;
  int m0 = blockIdx.x * 128 + wave * 16;
  v8f acc[8] = {};
  for (int k0 = 0; k0 < XQ_COLS; k0 += 32) {
    // prefetch next A chunk into cache hierarchy (global_prefetch_b8)
    if (k0 + 32 < XQ_COLS)
      __builtin_prefetch(xq + (size_t)m0 * XQ_COLS + k0 + 32, 0, 0);
    v16bf a = load_a_bf16(xq, XQ_COLS, m0, k0);
#pragma unroll
    for (int j = 0; j < 8; ++j) {
      v16bf b = load_bT(winvT, XQ_COLS, j * 16, k0);
      acc[j] = wmma_bf16(a, b, acc[j]);
    }
  }
  int lane = threadIdx.x & 31;
#pragma unroll
  for (int j = 0; j < 8; ++j) {
    int col = j * 16 + (lane & 15);
    float bias = b_out[col];
#pragma unroll
    for (int i = 0; i < 8; ++i) {
      int row = m0 + i + ((lane >> 4) ? 8 : 0);
      out[(size_t)row * DIMD + col] = acc[j][i] + bias;
    }
  }
}

// ---------------------------------------------------------------------------
extern "C" void kernel_launch(void* const* d_in, const int* in_sizes, int n_in,
                              void* d_out, int out_size, void* d_ws, size_t ws_size,
                              hipStream_t stream) {
  (void)in_sizes; (void)n_in; (void)out_size; (void)ws_size;
  const float* x      = (const float*)d_in[0];
  const float* w_qkv  = (const float*)d_in[1];
  const float* w_out  = (const float*)d_in[2];
  const float* b_out  = (const float*)d_in[3];
  float* out = (float*)d_out;

  char* ws = (char*)d_ws;
  size_t off = 0;
  auto alloc = [&](size_t bytes) -> void* {
    void* p = ws + off;
    off = (off + bytes + 255) & ~(size_t)255;
    return p;
  };
  bf16_t* wftT    = (bf16_t*)alloc((size_t)FT_COLS * DIMD * 2);            // 1.2 MB
  bf16_t* winvT   = (bf16_t*)alloc((size_t)DIMD * XQ_COLS * 2);            // 0.4 MB
  bf16_t* xbf     = (bf16_t*)alloc((size_t)MM * DIMD * 2);                 // 8.4 MB
  bf16_t* ft      = (bf16_t*)alloc((size_t)MM * FT_COLS * 2);              // 302 MB
  bf16_t* qkT     = (bf16_t*)alloc((size_t)2 * BB * HEADS * PF * NN * 2);  // 101 MB
  float*  att_raw = (float*) alloc((size_t)BB * HEADS * PF * PF * 4);      // 4.7 MB
  bf16_t* att     = (bf16_t*)alloc((size_t)BB * HEADS * PF * PF * 2);      // 2.4 MB
  bf16_t* xq      = (bf16_t*)alloc((size_t)MM * XQ_COLS * 2);              // 101 MB

  k_fold_wft <<<(FT_COLS * DIMD + 255) / 256, 256, 0, stream>>>(w_qkv, wftT);
  k_fold_winv<<<(DIMD * XQ_COLS + 255) / 256, 256, 0, stream>>>(w_out, winvT);
  k_x2bf     <<<(MM * DIMD + 255) / 256, 256, 0, stream>>>(x, xbf);

  k_gemm_ft<<<dim3(MM / 128, FT_COLS / 64), 256, 0, stream>>>(xbf, wftT, ft);

  size_t abs_total = (size_t)2 * BB * HEADS * PF * NN;
  k_abs_t<<<(unsigned)((abs_total + 255) / 256), 256, 0, stream>>>(ft, qkT);

  k_gemm_att<<<BB * HEADS * 6, 32, 0, stream>>>(qkT, att_raw);
  k_softmax <<<BB * HEADS * PF, 128, 0, stream>>>(att_raw, att);

  k_gemm_xqkv<<<16 * 2 * HEADS * BB, 256, 0, stream>>>(ft, att, xq);
  k_gemm_out <<<MM / 128, 256, 0, stream>>>(xq, winvT, b_out, out);
}
